// GNN_5772436046248
// MI455X (gfx1250) — matrix operand order repaired and checked by reading, not compiled, not run on hardware
//
#include <hip/hip_runtime.h>
#include <hip/hip_bf16.h>

// ---------------------------------------------------------------------------
// MPNN (NNConv + GRU + Set2Set) for MI455X (gfx1250, wave32, WMMA).
// Key design: the per-edge 64x64 weight W_edge = e1 @ w1c.T is NEVER
// materialized (819 MB f32 -> ~3.3 GB HBM traffic). Instead message passing
// is fused as  m[e,o] = sum_{h,k} hsrc[e,h]*e1[e,k]*w1c[h*64+o,k]
// = (outer-product A tile) x (constant 512KB f16 B matrix, L2 resident),
// executed entirely with v_wmma_f32_16x16x32_f16.
// ---------------------------------------------------------------------------

#define HH   64
#define NNODE 10000
#define NEDGE 50000
#define NBATCH 256
#define NFEAT 11
#define NTGT  12

typedef __attribute__((ext_vector_type(16))) _Float16 v16h;
typedef __attribute__((ext_vector_type(8)))  float    v8f;

__device__ __forceinline__ int lane_id() { return (int)(threadIdx.x & 31u); }

// A-fragment (16x32, f16) from f32 row-major source, lda in elements.
__device__ __forceinline__ v16h ld_a_f32(const float* A, int lda) {
  const int l = lane_id();
  const float* p = A + (l & 15) * lda + ((l >> 4) * 16);
  v16h a;
#pragma unroll
  for (int i = 0; i < 8; ++i) { a[i] = (_Float16)p[i]; a[8 + i] = (_Float16)p[8 + i]; }
  return a;
}

// A-fragment (16x32, f16) from f16 row-major source.
__device__ __forceinline__ v16h ld_a_f16(const _Float16* A, int lda) {
  const int l = lane_id();
  const _Float16* p = A + (l & 15) * lda + ((l >> 4) * 16);
  v16h a;
#pragma unroll
  for (int i = 0; i < 8; ++i) { a[i] = p[i]; a[8 + i] = p[8 + i]; }
  return a;
}

// B-fragment (32x16, f16) from packed [K][ldb] f16 matrix (caller pre-offsets
// to (k0, col0)).
__device__ __forceinline__ v16h ld_b(const _Float16* Bp, int ldb) {
  const int l = lane_id();
  const _Float16* p = Bp + ((l >> 4) * 16) * ldb + (l & 15);
  v16h b;
#pragma unroll
  for (int i = 0; i < 16; ++i) b[i] = p[i * ldb];
  return b;
}

__device__ __forceinline__ v8f wmma16(v16h a, v16h b, v8f c) {
  return __builtin_amdgcn_wmma_f32_16x16x32_f16(false, a, false, b, (short)0, c,
                                                false, false);
}

__device__ __forceinline__ float sigm(float x) { return 1.f / (1.f + __expf(-x)); }

__device__ __forceinline__ unsigned fkey(float f) {
  unsigned u = __float_as_uint(f);
  return (u & 0x80000000u) ? ~u : (u | 0x80000000u);
}
__device__ __forceinline__ float fdec(unsigned k) {
  return (k & 0x80000000u) ? __uint_as_float(k ^ 0x80000000u)
                           : __uint_as_float(~k);
}

// ---------------------------------------------------------------------------
// Generic WMMA GEMM: D[rows,outs] = act(A[rows,K] * Bp[K,outs] + bias)
// Bp is pre-packed f16 [K][ldb] with ldb = outs rounded up to 16 (zero fill).
// Block = 128 threads (4 waves), each wave one 16x16 tile; grid.y covers 64
// columns per block. rows must be a multiple of 16 (true for all our GEMMs).
// ---------------------------------------------------------------------------
template <int ACT, int ACC, int OUT16>
__global__ void __launch_bounds__(128) k_gemm(const float* __restrict__ A, int lda,
                                              const _Float16* __restrict__ Bp, int ldb,
                                              const float* __restrict__ bias,
                                              void* __restrict__ Dst, int ldd,
                                              int K, int outs) {
  const int row0 = blockIdx.x * 16;
  const int col0 = blockIdx.y * 64 + (int)(threadIdx.x >> 5) * 16;
  if (col0 >= ldb) return;  // wave-uniform
  v8f acc = {};
  for (int k0 = 0; k0 < K; k0 += 32) {
    v16h a = ld_a_f32(A + row0 * lda + k0, lda);
    v16h b = ld_b(Bp + k0 * ldb + col0, ldb);
    acc = wmma16(a, b, acc);
  }
  const int l = lane_id();
  const int n = col0 + (l & 15);
  const int rb = (l >> 4) * 8;
  if (n < outs) {
    const float bv = bias[n];
#pragma unroll
    for (int r = 0; r < 8; ++r) {
      float v = acc[r] + bv;
      if (ACT) v = v > 0.f ? v : 0.f;
      const int idx = (row0 + rb + r) * ldd + n;
      if (OUT16)      ((_Float16*)Dst)[idx] = (_Float16)v;
      else if (ACC)   ((float*)Dst)[idx] += v;
      else            ((float*)Dst)[idx] = v;
    }
  }
}

// ---------------------------------------------------------------------------
// Fused NNConv message pass for one tile of 16 edges.
//   m[e,o] = sum_h hsrc[e,h] * ( b1c[h*64+o] + sum_k e1[e,k]*w1c[h*64+o,k] )
// A-fragments are e1 fragments scaled per 32-chunk by the scalar hsrc[e,h]
// (h is constant inside each 32-wide K' chunk). B comes from Wpp, a
// fragment-linear f16 pack of w1c (512 KB, L2 resident).
// The h loop is unrolled x2 with 4 disjoint product register sets so the
// v_pk_mul_f16 scaling never WARs against an in-flight WMMA's A operand
// (avoids the 4-coexec-NOP WMMA->VALU hazard penalty).
// ---------------------------------------------------------------------------
__global__ void __launch_bounds__(128) k_nnconv(const _Float16* __restrict__ e1,
                                                const v16h* __restrict__ Wpp,
                                                const _Float16* __restrict__ b1c16,
                                                const float* __restrict__ h,
                                                const int* __restrict__ src,
                                                const int* __restrict__ dst,
                                                float* __restrict__ m) {
  __shared__ _Float16 hs[16 * 64];
  const int e0 = blockIdx.x * 16;
  for (int t = threadIdx.x; t < 16 * 64; t += 128) {
    const int er = t >> 6, c = t & 63;
    hs[t] = (_Float16)h[src[e0 + er] * 64 + c];
  }
  __syncthreads();

  const int w = (int)(threadIdx.x >> 5);
  const int l = lane_id();
  const int row = l & 15;

  const v16h a0 = ld_a_f16(e1 + e0 * 64, 64);        // e1 k=0..31
  const v16h a1 = ld_a_f16(e1 + e0 * 64 + 32, 64);   // e1 k=32..63

  v8f acc = {};
  // bias b1c contribution: hsrc[16x64] @ Bc[64x64]  (Bc == b1c reshaped)
  {
    v16h ha0 = ld_a_f16(hs, 64);
    v16h ha1 = ld_a_f16(hs + 32, 64);
    v16h b0 = ld_b(b1c16 + 0 * 64 + w * 16, 64);
    v16h b1 = ld_b(b1c16 + 32 * 64 + w * 16, 64);
    acc = wmma16(ha0, b0, acc);
    acc = wmma16(ha1, b1, acc);
  }

  const v16h* wp = Wpp + w * 32 + l;  // fragment stream for this wave's o-tile
  const _Float16* hrow = hs + row * 64;
#pragma unroll 2
  for (int hh = 0; hh < 64; hh += 2) {
    // two adjacent hsrc scalars (one aligned ds_load_b32)
    const _Float16 s0 = hrow[hh];
    const _Float16 s1 = hrow[hh + 1];
    v16h p00, p01, p10, p11;  // 4 disjoint product sets
#pragma unroll
    for (int i = 0; i < 16; ++i) {
      p00[i] = a0[i] * s0; p01[i] = a1[i] * s0;
      p10[i] = a0[i] * s1; p11[i] = a1[i] * s1;
    }
    if (hh + 4 < 64)
      __builtin_prefetch((const void*)(wp + (hh * 2 + 8) * 128), 0, 0);
    v16h b0 = wp[(hh * 2 + 0) * 128];
    v16h b1 = wp[(hh * 2 + 1) * 128];
    v16h b2 = wp[(hh * 2 + 2) * 128];
    v16h b3 = wp[(hh * 2 + 3) * 128];
    acc = wmma16(p00, b0, acc);
    acc = wmma16(p01, b1, acc);
    acc = wmma16(p10, b2, acc);
    acc = wmma16(p11, b3, acc);
  }

  // scatter-add into m[dst]
  const int o = w * 16 + (l & 15);
  const int rb = (l >> 4) * 8;
#pragma unroll
  for (int r = 0; r < 8; ++r) {
    const int e = e0 + rb + r;
    atomicAdd(&m[dst[e] * 64 + o], acc[r]);
  }
}

// ---------------------------------------------------------------------------
// Small VALU kernels
// ---------------------------------------------------------------------------
__global__ void k_zero(float* p, int n) {
  int i = blockIdx.x * blockDim.x + threadIdx.x;
  if (i < n) p[i] = 0.f;
}

__global__ void k_init_h(const float* __restrict__ x, float* __restrict__ h) {
  int i = blockIdx.x * blockDim.x + threadIdx.x;
  if (i >= NNODE * 64) return;
  int n = i >> 6, c = i & 63;
  h[i] = (c < NFEAT) ? x[n * NFEAT + c] : 0.f;
}

// pack B: Bp[k*ldb + j] = W[j*K + k]  (zero fill j >= outs)
__global__ void k_pack_b(const float* __restrict__ W, _Float16* __restrict__ Bp,
                         int K, int outs, int ldb) {
  int i = blockIdx.x * blockDim.x + threadIdx.x;
  if (i >= K * ldb) return;
  int k = i / ldb, j = i % ldb;
  Bp[i] = (_Float16)(j < outs ? W[j * K + k] : 0.f);
}

// fragment-linear pack of w1c[4096,64] for k_nnconv
__global__ void k_pack_wpp(const float* __restrict__ w1c, _Float16* __restrict__ Wpp) {
  int i = blockIdx.x * blockDim.x + threadIdx.x;
  if (i >= 128 * 4 * 32 * 16) return;
  const int ii = i & 15;
  const int lane = (i >> 4) & 31;
  const int ot = (i >> 9) & 3;
  const int c = i >> 11;            // 0..127  (= h*2 + half)
  const int h = c >> 1, half = c & 1;
  const int klocal = (lane >> 4) * 16 + ii;   // 0..31
  const int n = lane & 15;
  const int j = h * 64 + ot * 16 + n;         // w1c row (= h*64 + o)
  const int k = half * 32 + klocal;           // w1c col
  Wpp[i] = (_Float16)w1c[j * 64 + k];
}

__global__ void k_f32_to_f16(const float* s, _Float16* d, int n) {
  int i = blockIdx.x * blockDim.x + threadIdx.x;
  if (i < n) d[i] = (_Float16)s[i];
}

// edge MLP layer a: e1a[e,c] = relu(b1a[c] + edge_attr[e,:4] . w1a[c,:4])
__global__ void k_edge_a(const float* __restrict__ ea, const float* __restrict__ w1a,
                         const float* __restrict__ b1a, float* __restrict__ out) {
  int i = blockIdx.x * blockDim.x + threadIdx.x;
  if (i >= NEDGE * 64) return;
  int e = i >> 6, c = i & 63;
  const float* a = ea + e * 4;
  const float* w = w1a + c * 4;
  float v = b1a[c] + a[0] * w[0] + a[1] * w[1] + a[2] * w[2] + a[3] * w[3];
  out[i] = v > 0.f ? v : 0.f;
}

__global__ void k_gru_gate(const float* __restrict__ gi, const float* __restrict__ gh,
                           float* __restrict__ h) {
  int i = blockIdx.x * blockDim.x + threadIdx.x;
  if (i >= NNODE * 64) return;
  int n = i >> 6, c = i & 63;
  const float* Gi = gi + n * 192;
  const float* Gh = gh + n * 192;
  float r = sigm(Gi[c] + Gh[c]);
  float z = sigm(Gi[64 + c] + Gh[64 + c]);
  float nn = tanhf(Gi[128 + c] + r * Gh[128 + c]);
  h[i] = (1.f - z) * nn + z * h[i];
}

__global__ void k_lstm_gate(const float* __restrict__ g, float* __restrict__ hcur,
                            float* __restrict__ ccur) {
  int i = blockIdx.x * blockDim.x + threadIdx.x;
  if (i >= NBATCH * 128) return;
  int b = i >> 7, c = i & 127;
  const float* G = g + b * 512;
  float ig = sigm(G[c]);
  float fg = sigm(G[128 + c]);
  float gg = tanhf(G[256 + c]);
  float og = sigm(G[384 + c]);
  float cn = fg * ccur[i] + ig * gg;
  ccur[i] = cn;
  hcur[i] = og * tanhf(cn);
}

// lin: h2[n,j] = lin_b[j] + [h,x][n,:] . lin_w[j,:75]
__global__ void k_lin(const float* __restrict__ h, const float* __restrict__ x,
                      const float* __restrict__ lw, const float* __restrict__ lb,
                      float* __restrict__ h2) {
  int i = blockIdx.x * blockDim.x + threadIdx.x;
  if (i >= NNODE * 128) return;
  int n = i >> 7, j = i & 127;
  const float* w = lw + j * 75;
  float v = lb[j];
#pragma unroll
  for (int k = 0; k < 64; ++k) v += h[n * 64 + k] * w[k];
#pragma unroll
  for (int k = 0; k < NFEAT; ++k) v += x[n * NFEAT + k] * w[64 + k];
  h2[i] = v;
}

// attention score: e[n] = h2[n,:] . q[batch[n],:], segment max via keyed atomic
__global__ void __launch_bounds__(128) k_attn_e(const float* __restrict__ h2,
                                                const float* __restrict__ q,
                                                const int* __restrict__ batch,
                                                float* __restrict__ e,
                                                unsigned* __restrict__ emaxk) {
  int node = blockIdx.x * 4 + (int)(threadIdx.x >> 5);
  if (node >= NNODE) return;
  int l = lane_id();
  int b = batch[node];
  const float* hr = h2 + node * 128;
  const float* qr = q + b * 128;
  float s = hr[l] * qr[l] + hr[32 + l] * qr[32 + l] + hr[64 + l] * qr[64 + l] +
            hr[96 + l] * qr[96 + l];
#pragma unroll
  for (int off = 16; off; off >>= 1) s += __shfl_xor(s, off, 32);
  if (l == 0) {
    e[node] = s;
    atomicMax(&emaxk[b], fkey(s));
  }
}

__global__ void k_attn_a(const float* __restrict__ e, const int* __restrict__ batch,
                         const unsigned* __restrict__ emaxk, float* __restrict__ a,
                         float* __restrict__ denom) {
  int n = blockIdx.x * blockDim.x + threadIdx.x;
  if (n >= NNODE) return;
  int b = batch[n];
  float v = __expf(e[n] - fdec(emaxk[b]));
  a[n] = v;
  atomicAdd(&denom[b], v);
}

__global__ void k_attn_r(const float* __restrict__ a, const float* __restrict__ denom,
                         const int* __restrict__ batch, const float* __restrict__ h2,
                         float* __restrict__ r) {
  int i = blockIdx.x * blockDim.x + threadIdx.x;
  if (i >= NNODE * 128) return;
  int n = i >> 7, c = i & 127;
  int b = batch[n];
  float an = a[n] / denom[b];
  atomicAdd(&r[b * 128 + c], an * h2[i]);
}

__global__ void k_qstar(const float* __restrict__ q, const float* __restrict__ r,
                        float* __restrict__ qs) {
  int i = blockIdx.x * blockDim.x + threadIdx.x;
  if (i >= NBATCH * 256) return;
  int b = i >> 8, c = i & 255;
  qs[i] = (c < 128) ? q[b * 128 + c] : r[b * 128 + (c - 128)];
}

// ---------------------------------------------------------------------------
extern "C" void kernel_launch(void* const* d_in, const int* in_sizes, int n_in,
                              void* d_out, int out_size, void* d_ws, size_t ws_size,
                              hipStream_t stream) {
  const float* x   = (const float*)d_in[0];
  const float* ea  = (const float*)d_in[1];
  const int* eidx  = (const int*)d_in[2];
  const int* batch = (const int*)d_in[3];
  const float* w1a = (const float*)d_in[5];
  const float* b1a = (const float*)d_in[6];
  const float* w1b = (const float*)d_in[7];
  const float* b1b = (const float*)d_in[8];
  const float* w1c = (const float*)d_in[9];
  const float* b1c = (const float*)d_in[10];
  const float* gwih = (const float*)d_in[11];
  const float* gwhh = (const float*)d_in[12];
  const float* gbih = (const float*)d_in[13];
  const float* gbhh = (const float*)d_in[14];
  const float* linw = (const float*)d_in[15];
  const float* linb = (const float*)d_in[16];
  const float* wih0 = (const float*)d_in[17];
  const float* whh0 = (const float*)d_in[18];
  const float* bih0 = (const float*)d_in[19];
  const float* bhh0 = (const float*)d_in[20];
  const float* wih1 = (const float*)d_in[21];
  const float* whh1 = (const float*)d_in[22];
  const float* bih1 = (const float*)d_in[23];
  const float* bhh1 = (const float*)d_in[24];
  const float* w2a = (const float*)d_in[25];
  const float* b2a = (const float*)d_in[26];
  const float* w2b = (const float*)d_in[27];
  const float* b2b = (const float*)d_in[28];
  const int* src = eidx;
  const int* dst = eidx + NEDGE;

  // ---- workspace carve-up (256B aligned) ----
  char* base = (char*)d_ws;
  size_t off = 0;
  auto alloc = [&](size_t bytes) -> void* {
    off = (off + 255) & ~(size_t)255;
    void* p = base + off;
    off += bytes;
    return p;
  };
  float*    e1a   = (float*)alloc((size_t)NEDGE * 64 * 4);
  _Float16* e1h   = (_Float16*)alloc((size_t)NEDGE * 64 * 2);
  float*    hbuf  = (float*)alloc((size_t)NNODE * 64 * 4);
  float*    mbuf  = (float*)alloc((size_t)NNODE * 64 * 4);
  float*    gi    = (float*)alloc((size_t)NNODE * 192 * 4);
  float*    gh    = (float*)alloc((size_t)NNODE * 192 * 4);
  float*    h2    = (float*)alloc((size_t)NNODE * 128 * 4);
  _Float16* wpp   = (_Float16*)alloc((size_t)128 * 4 * 32 * 16 * 2);
  _Float16* b1c16 = (_Float16*)alloc((size_t)4096 * 2);
  _Float16* w1bp  = (_Float16*)alloc((size_t)64 * 64 * 2);
  _Float16* gwihp = (_Float16*)alloc((size_t)64 * 192 * 2);
  _Float16* gwhhp = (_Float16*)alloc((size_t)64 * 192 * 2);
  _Float16* wih0p = (_Float16*)alloc((size_t)256 * 512 * 2);
  _Float16* whh0p = (_Float16*)alloc((size_t)128 * 512 * 2);
  _Float16* wih1p = (_Float16*)alloc((size_t)128 * 512 * 2);
  _Float16* whh1p = (_Float16*)alloc((size_t)128 * 512 * 2);
  _Float16* w2ap  = (_Float16*)alloc((size_t)256 * 512 * 2);
  _Float16* w2bp  = (_Float16*)alloc((size_t)512 * 16 * 2);
  float*    qstar = (float*)alloc((size_t)NBATCH * 256 * 4);
  float*    hc    = (float*)alloc((size_t)4 * NBATCH * 128 * 4);
  float*    g0    = (float*)alloc((size_t)NBATCH * 512 * 4);
  float*    g1    = (float*)alloc((size_t)NBATCH * 512 * 4);
  float*    esc   = (float*)alloc((size_t)NNODE * 4);
  float*    asc   = (float*)alloc((size_t)NNODE * 4);
  float*    att   = (float*)alloc((size_t)(256 + 256 + NBATCH * 128) * 4);
  unsigned* emaxk = (unsigned*)att;
  float*    denom = att + 256;
  float*    rbuf  = att + 512;
  float* h0 = hc, *c0 = hc + NBATCH * 128, *h1 = hc + 2 * NBATCH * 128,
       * c1 = hc + 3 * NBATCH * 128;

  const int TB = 256;
  auto blk = [&](int n) { return dim3((unsigned)((n + TB - 1) / TB)); };

  // ---- weight packing (f16, WMMA B layout) ----
  k_pack_b<<<blk(64 * 64), TB, 0, stream>>>(w1b, w1bp, 64, 64, 64);
  k_pack_b<<<blk(64 * 192), TB, 0, stream>>>(gwih, gwihp, 64, 192, 192);
  k_pack_b<<<blk(64 * 192), TB, 0, stream>>>(gwhh, gwhhp, 64, 192, 192);
  k_pack_b<<<blk(256 * 512), TB, 0, stream>>>(wih0, wih0p, 256, 512, 512);
  k_pack_b<<<blk(128 * 512), TB, 0, stream>>>(whh0, whh0p, 128, 512, 512);
  k_pack_b<<<blk(128 * 512), TB, 0, stream>>>(wih1, wih1p, 128, 512, 512);
  k_pack_b<<<blk(128 * 512), TB, 0, stream>>>(whh1, whh1p, 128, 512, 512);
  k_pack_b<<<blk(256 * 512), TB, 0, stream>>>(w2a, w2ap, 256, 512, 512);
  k_pack_b<<<blk(512 * 16), TB, 0, stream>>>(w2b, w2bp, 512, 12, 16);
  k_pack_wpp<<<blk(128 * 4 * 32 * 16), TB, 0, stream>>>(w1c, wpp);
  k_f32_to_f16<<<blk(4096), TB, 0, stream>>>(b1c, b1c16, 4096);

  // ---- node/edge prep ----
  k_init_h<<<blk(NNODE * 64), TB, 0, stream>>>(x, hbuf);
  k_edge_a<<<blk(NEDGE * 64), TB, 0, stream>>>(ea, w1a, b1a, e1a);
  // edge MLP layer b (relu, f16 out): e1h = relu(e1a @ w1b.T + b1b)
  k_gemm<1, 0, 1><<<dim3(NEDGE / 16, 1), 128, 0, stream>>>(
      e1a, 64, w1bp, 64, b1b, (void*)e1h, 64, 64, 64);

  // ---- T=3 rounds of fused NNConv + GRU ----
  for (int t = 0; t < 3; ++t) {
    k_zero<<<blk(NNODE * 64), TB, 0, stream>>>(mbuf, NNODE * 64);
    k_nnconv<<<dim3(NEDGE / 16), 128, 0, stream>>>(e1h, (const v16h*)wpp, b1c16,
                                                   hbuf, src, dst, mbuf);
    k_gemm<0, 0, 0><<<dim3(NNODE / 16, 3), 128, 0, stream>>>(
        mbuf, 64, gwihp, 192, gbih, (void*)gi, 192, 64, 192);
    k_gemm<0, 0, 0><<<dim3(NNODE / 16, 3), 128, 0, stream>>>(
        hbuf, 64, gwhhp, 192, gbhh, (void*)gh, 192, 64, 192);
    k_gru_gate<<<blk(NNODE * 64), TB, 0, stream>>>(gi, gh, hbuf);
  }

  // ---- lin: h2 = [h, x] @ lin_w.T + lin_b ----
  k_lin<<<blk(NNODE * 128), TB, 0, stream>>>(hbuf, x, linw, linb, h2);

  // ---- Set2Set readout, M=3 ----
  k_zero<<<blk(NBATCH * 256), TB, 0, stream>>>(qstar, NBATCH * 256);
  k_zero<<<blk(4 * NBATCH * 128), TB, 0, stream>>>(hc, 4 * NBATCH * 128);
  for (int it = 0; it < 3; ++it) {
    // LSTM layer 0: g0 = q_star@wih0.T + bih0 + h0@whh0.T + bhh0
    k_gemm<0, 0, 0><<<dim3(NBATCH / 16, 8), 128, 0, stream>>>(
        qstar, 256, wih0p, 512, bih0, (void*)g0, 512, 256, 512);
    k_gemm<0, 1, 0><<<dim3(NBATCH / 16, 8), 128, 0, stream>>>(
        h0, 128, whh0p, 512, bhh0, (void*)g0, 512, 128, 512);
    k_lstm_gate<<<blk(NBATCH * 128), TB, 0, stream>>>(g0, h0, c0);
    // LSTM layer 1
    k_gemm<0, 0, 0><<<dim3(NBATCH / 16, 8), 128, 0, stream>>>(
        h0, 128, wih1p, 512, bih1, (void*)g1, 512, 128, 512);
    k_gemm<0, 1, 0><<<dim3(NBATCH / 16, 8), 128, 0, stream>>>(
        h1, 128, whh1p, 512, bhh1, (void*)g1, 512, 128, 512);
    k_lstm_gate<<<blk(NBATCH * 128), TB, 0, stream>>>(g1, h1, c1);
    // attention
    k_zero<<<blk(512 + NBATCH * 128), TB, 0, stream>>>(att, 512 + NBATCH * 128);
    k_attn_e<<<dim3((NNODE + 3) / 4), 128, 0, stream>>>(h2, h1, batch, esc, emaxk);
    k_attn_a<<<blk(NNODE), TB, 0, stream>>>(esc, batch, emaxk, asc, denom);
    k_attn_r<<<blk(NNODE * 128), TB, 0, stream>>>(asc, denom, batch, h2, rbuf);
    k_qstar<<<blk(NBATCH * 256), TB, 0, stream>>>(h1, rbuf, qstar);
  }

  // ---- nn2 readout: out = relu(q_star@w2a.T+b2a) @ w2b.T + b2b ----
  k_gemm<1, 0, 0><<<dim3(NBATCH / 16, 8), 128, 0, stream>>>(
      qstar, 256, w2ap, 512, b2a, (void*)g0, 512, 256, 512);
  k_gemm<0, 0, 0><<<dim3(NBATCH / 16, 1), 128, 0, stream>>>(
      g0, 512, w2bp, 16, b2b, d_out, NTGT, 512, 12);
}
